// HeadALiBi_49194555408424
// MI455X (gfx1250) — matrix operand order repaired and checked by reading, not compiled, not run on hardware
//
#include <hip/hip_runtime.h>
#include <hip/hip_bf16.h>

typedef __attribute__((ext_vector_type(16))) _Float16 v16h;
typedef __attribute__((ext_vector_type(8)))  _Float16 v8h;
typedef __attribute__((ext_vector_type(4)))  _Float16 v4h;
typedef __attribute__((ext_vector_type(2)))  _Float16 v2h;
typedef __attribute__((ext_vector_type(8)))  float    v8f;
typedef __attribute__((ext_vector_type(4)))  float    v4f;

constexpr int NB = 8192;   // batches
constexpr int TT = 32;     // sequence length
constexpr int CC = 64;     // model dim
constexpr int HH = 64;     // head size
constexpr int WPB = 2;     // waves per block (1 batch per wave)
// LDS: shared f16 weights (3 x 64x64) + per-wave {q|P, k, vT}
constexpr int W_LDS   = 3 * CC * HH;             // 12288 halves
constexpr int PW_LDS  = 2048 + 2048 + 2048;      // q/P, k, vT halves

__device__ __forceinline__ v8f wmma_f16(v16h a, v16h b, v8f c) {
  // D = A(16x32 f16) x B(32x16 f16) + C(16x16 f32)
  return __builtin_amdgcn_wmma_f32_16x16x32_f16(false, a, false, b, (short)0, c,
                                                false, false);
}

// A fragment (16x32 f16) from f32 row-major global: row m = lane%16,
// K runs [c0..c0+7] and [c0+16..c0+23], c0 = kbase + 8*(lane/16).
__device__ __forceinline__ v16h afrag_f32g(const float* __restrict__ base,
                                           int row, int c0, int stride) {
  const v4f* p = reinterpret_cast<const v4f*>(base + (size_t)row * stride + c0);
  v4f f0 = p[0], f1 = p[1], f2 = p[4], f3 = p[5];
  v16h a;
#pragma unroll
  for (int i = 0; i < 4; ++i) {
    a[i]      = (_Float16)f0[i];
    a[4 + i]  = (_Float16)f1[i];
    a[8 + i]  = (_Float16)f2[i];
    a[12 + i] = (_Float16)f3[i];
  }
  return a;
}

// A fragment from f16 LDS (row-major, stride halves).
__device__ __forceinline__ v16h afrag_lds(const _Float16* base, int row,
                                          int c0, int stride) {
  const v8h* p = reinterpret_cast<const v8h*>(base + row * stride + c0);
  v8h lo = p[0], hi = p[2];   // +0 and +16 halves
  v16h a;
#pragma unroll
  for (int i = 0; i < 8; ++i) { a[i] = lo[i]; a[8 + i] = hi[i]; }
  return a;
}

// B fragment from f16 LDS holding B^T row-major (row n contiguous in K):
// column n = lane%16, K runs [k0..k0+7] and [k0+16..k0+23], k0 = kbase + 8*(lane/16).
__device__ __forceinline__ v16h bfrag_lds(const _Float16* base, int n,
                                          int k0, int stride) {
  const v8h* p = reinterpret_cast<const v8h*>(base + n * stride + k0);
  v8h lo = p[0], hi = p[2];   // +0 and +16 halves
  v16h b;
#pragma unroll
  for (int i = 0; i < 8; ++i) { b[i] = lo[i]; b[8 + i] = hi[i]; }
  return b;
}

__global__ __launch_bounds__(WPB * 32)
void head_alibi_kernel(const float* __restrict__ x,
                       const float* __restrict__ Wk,
                       const float* __restrict__ Wq,
                       const float* __restrict__ Wv,
                       float* __restrict__ out,
                       float* __restrict__ attn) {
  __shared__ __align__(16) _Float16 lds[W_LDS + WPB * PW_LDS];  // 49152 B

  const int tid  = threadIdx.x;
  const int wave = tid >> 5;
  const int lane = tid & 31;
  const int hw   = lane >> 4;   // half-wave group (0..1)
  const int ln   = lane & 15;   // lane within group
  const int b    = blockIdx.x * WPB + wave;

  // ---- block-cooperative weight staging: f32 global -> f16 LDS ----
  _Float16* wsh = lds;                       // [3][64][64], W[h][c] row-major
  {
    const float* Wsrc[3] = {Wq, Wk, Wv};
#pragma unroll
    for (int pj = 0; pj < 3; ++pj) {
      const float* W = Wsrc[pj];
      _Float16* dst = wsh + pj * (CC * HH);
#pragma unroll
      for (int it = 0; it < 16; ++it) {      // 16 * 64 lanes * 4 = 4096
        int idx = (it * (WPB * 32) + tid) * 4;
        v4f w = *reinterpret_cast<const v4f*>(W + idx);
        v4h h = {(_Float16)w[0], (_Float16)w[1], (_Float16)w[2], (_Float16)w[3]};
        *reinterpret_cast<v4h*>(dst + idx) = h;
      }
    }
  }
  __syncthreads();

  _Float16* qsh = lds + W_LDS + wave * PW_LDS;  // q [32][64]; later reused as P
  _Float16* ksh = qsh + TT * CC;                // k [32][64] row-major
  _Float16* vth = ksh + TT * CC;                // v^T [64][32]
  _Float16* psh = qsh;                          // P [32][32] overlays q

  const float* xb = x + (size_t)b * TT * CC;

  // ---- x A-fragments (reused by all three projections) ----
  v16h xa[2][2];
#pragma unroll
  for (int mt = 0; mt < 2; ++mt)
#pragma unroll
    for (int kk = 0; kk < 2; ++kk)
      xa[mt][kk] = afrag_f32g(xb, 16 * mt + ln, 32 * kk + 8 * hw, CC);

  // ---- projections: q, k (row-major), v (transposed) ----
#pragma unroll
  for (int pj = 0; pj < 3; ++pj) {
    const _Float16* W = wsh + pj * (CC * HH);
#pragma unroll
    for (int nt = 0; nt < 4; ++nt) {
      v16h b0 = bfrag_lds(W, 16 * nt + ln, 8 * hw, CC);        // K 0..31
      v16h b1 = bfrag_lds(W, 16 * nt + ln, 32 + 8 * hw, CC);   // K 32..63
#pragma unroll
      for (int mt = 0; mt < 2; ++mt) {
        v8f acc = {};
        acc = wmma_f16(xa[mt][0], b0, acc);
        acc = wmma_f16(xa[mt][1], b1, acc);
        if (pj == 2) {       // v stored transposed: vT[h][s], packed pair stores
#pragma unroll
          for (int j = 0; j < 8; j += 2) {
            v2h t = {(_Float16)acc[j], (_Float16)acc[j + 1]};
            *reinterpret_cast<v2h*>(
                &vth[(16 * nt + ln) * TT + 16 * mt + j + 8 * hw]) = t;
          }
        } else {             // q / k row-major [t][h]
          _Float16* dst = (pj == 0) ? qsh : ksh;
#pragma unroll
          for (int j = 0; j < 8; ++j)
            dst[(16 * mt + j + 8 * hw) * CC + 16 * nt + ln] = (_Float16)acc[j];
        }
      }
    }
  }

  asm volatile("s_wait_dscnt 0" ::: "memory");

  // ---- scores = q @ k^T : [32,32], K = 64 ----
  v16h qa[2][2];
#pragma unroll
  for (int mt = 0; mt < 2; ++mt) {
    qa[mt][0] = afrag_lds(qsh, 16 * mt + ln, 8 * hw, CC);
    qa[mt][1] = afrag_lds(qsh, 16 * mt + ln, 32 + 8 * hw, CC);
  }
  v8f sc[2][2];
#pragma unroll
  for (int nt = 0; nt < 2; ++nt) {
    v16h kb0 = bfrag_lds(ksh, 16 * nt + ln, 8 * hw, CC);
    v16h kb1 = bfrag_lds(ksh, 16 * nt + ln, 32 + 8 * hw, CC);
#pragma unroll
    for (int mt = 0; mt < 2; ++mt) {
      v8f acc = {};
      acc = wmma_f16(qa[mt][0], kb0, acc);
      acc = wmma_f16(qa[mt][1], kb1, acc);
      sc[mt][nt] = acc;
    }
  }

  // ---- softmax with ALiBi bias; write P to global (f32) and LDS (f16) ----
  // (q fragments are already in registers, so P may overwrite q's LDS.)
  const float scale = 0.125f;    // 64^-0.5
  const float slope = 0.0625f;   // 2^(-8/2) * (0+1)
  float* attn_b = attn + (size_t)b * TT * TT;
#pragma unroll
  for (int mt = 0; mt < 2; ++mt) {
#pragma unroll
    for (int j = 0; j < 8; ++j) {
      float v0 = sc[mt][0][j] * scale - slope * (float)ln;          // col ln
      float v1 = sc[mt][1][j] * scale - slope * (float)(16 + ln);   // col 16+ln
      float mx = fmaxf(v0, v1);
#pragma unroll
      for (int d = 1; d < 16; d <<= 1) mx = fmaxf(mx, __shfl_xor(mx, d, 16));
      float e0 = __expf(v0 - mx);
      float e1 = __expf(v1 - mx);
      float s = e0 + e1;
#pragma unroll
      for (int d = 1; d < 16; d <<= 1) s += __shfl_xor(s, d, 16);
      float r  = 1.0f / s;
      float p0 = e0 * r, p1 = e1 * r;
      int m = 16 * mt + j + 8 * hw;
      attn_b[(size_t)m * TT + ln]      = p0;
      attn_b[(size_t)m * TT + 16 + ln] = p1;
      psh[m * TT + ln]      = (_Float16)p0;
      psh[m * TT + 16 + ln] = (_Float16)p1;
    }
  }

  asm volatile("s_wait_dscnt 0" ::: "memory");

  // ---- out = P @ v : [32,64], K = 32 ----
  float* out_b = out + (size_t)b * TT * HH;
#pragma unroll
  for (int mt = 0; mt < 2; ++mt) {
    v16h pa = afrag_lds(psh, 16 * mt + ln, 8 * hw, TT);
#pragma unroll
    for (int nt = 0; nt < 4; ++nt) {
      v16h vb = bfrag_lds(vth, 16 * nt + ln, 8 * hw, TT);
      v8f acc = {};
      acc = wmma_f16(pa, vb, acc);
#pragma unroll
      for (int j = 0; j < 8; ++j)
        out_b[(size_t)(16 * mt + j + 8 * hw) * HH + 16 * nt + ln] = acc[j];
    }
  }
}

extern "C" void kernel_launch(void* const* d_in, const int* in_sizes, int n_in,
                              void* d_out, int out_size, void* d_ws, size_t ws_size,
                              hipStream_t stream) {
  const float* x  = (const float*)d_in[0];
  const float* Wk = (const float*)d_in[1];
  const float* Wq = (const float*)d_in[2];
  const float* Wv = (const float*)d_in[3];
  float* out  = (float*)d_out;                                  // [8192,32,64]
  float* attn = (float*)d_out + (size_t)NB * TT * HH;           // [8192,32,32]

  dim3 grid(NB / WPB);
  dim3 block(WPB * 32);
  hipLaunchKernelGGL(head_alibi_kernel, grid, block, 0, stream,
                     x, Wk, Wq, Wv, out, attn);
}
